// Encoder_Decoder_79276506349699
// MI455X (gfx1250) — matrix operand order repaired and checked by reading, not compiled, not run on hardware
//
#include <hip/hip_runtime.h>
#include <hip/hip_bf16.h>
#include <math.h>

#define HIDDEN   512
#define VOCAB    32000
#define BATCH    64
#define SRC_LEN  50
#define TGT_LEN  50
#define DEC_STEPS 49
#define DROWS    (DEC_STEPS * BATCH)      // 3136 decoder h rows
#define MTILES   (DROWS / 16)             // 196
#define TPW      5                        // N-tiles per wave in projection
#define WVS      8                        // waves per block in projection
#define CHUNK_T  (TPW * WVS)              // 40 N-tiles per chunk (640 cols)
#define NCHUNK   (VOCAB / 16 / CHUNK_T)   // 50 chunks

typedef __attribute__((ext_vector_type(16))) __bf16          v16bf;
typedef __attribute__((ext_vector_type(8)))  float           v8f;
typedef __attribute__((ext_vector_type(16))) unsigned short  v16u;
typedef __attribute__((ext_vector_type(8)))  unsigned short  v8u;

__device__ __forceinline__ unsigned short f2bf(float f) {
    unsigned int x = __float_as_uint(f);
    x += 0x7FFFu + ((x >> 16) & 1u);           // round-to-nearest-even
    return (unsigned short)(x >> 16);
}
__device__ __forceinline__ float bf2f(unsigned short u) {
    return __uint_as_float(((unsigned int)u) << 16);
}

// A-matrix fragment (16x32 bf16, ISA layout): lane ln=lane&15 holds row M=ln.
// elements 0..7  -> K = k0 + 8*half + j
// elements 8..15 -> K = k0 + 16 + 8*half + j
__device__ __forceinline__ v16bf load_a_bf(const unsigned short* rowbase, int k0, int half) {
    const v8u* p0 = (const v8u*)(rowbase + k0 + 8 * half);
    const v8u* p1 = (const v8u*)(rowbase + k0 + 8 * half + 16);
    v8u a0 = *p0, a1 = *p1;
    v16u u;
#pragma unroll
    for (int j = 0; j < 8; ++j) { u[j] = a0[j]; u[8 + j] = a1[j]; }
    return __builtin_bit_cast(v16bf, u);
}
// B-matrix fragment (32x16 bf16): lane ln holds column N=ln; elements 0..7 -> K=k0+8*half+j,
// elements 8..15 -> K=k0+16+8*half+j. rowbase = &W[col][0] (W stored (N,K) row-major = B^T).
__device__ __forceinline__ v16bf load_b_bf(const unsigned short* rowbase, int k0, int half) {
    const v8u* p0 = (const v8u*)(rowbase + k0 + 8 * half);
    const v8u* p1 = (const v8u*)(rowbase + k0 + 8 * half + 16);
    v8u a0 = *p0, a1 = *p1;
    v16u u;
#pragma unroll
    for (int j = 0; j < 8; ++j) { u[j] = a0[j]; u[8 + j] = a1[j]; }
    return __builtin_bit_cast(v16bf, u);
}

// ---------------------------------------------------------------------------
// One LSTM step. grid = (32 hidden-slices, 4 batch-tiles), block = 128 (4 waves).
// Wave g computes the 16x16 gate-g tile for (mtile, slice); pointwise cell
// update fused via LDS. All matrix operands are pre-converted bf16.
// ---------------------------------------------------------------------------
__global__ __launch_bounds__(128)
void lstm_step_kernel(const int* __restrict__ tok,              // 64 tokens (this step)
                      const unsigned short* __restrict__ emb,   // VOCAB x 512 bf16
                      const unsigned short* __restrict__ Wih,   // 2048 x 512 bf16
                      const unsigned short* __restrict__ Whh,   // 2048 x 512 bf16
                      const float* __restrict__ bih,
                      const float* __restrict__ bhh,
                      const unsigned short* __restrict__ hprev, // 64 x 512 bf16
                      unsigned short* __restrict__ hnext,       // 64 x 512 bf16
                      float* __restrict__ cstate,               // 64 x 512 f32
                      unsigned short* __restrict__ hdec)        // optional 64x512 bf16
{
    const int slice = blockIdx.x;         // hidden column tile 0..31
    const int mtile = blockIdx.y;         // batch tile 0..3
    const int gate  = threadIdx.x >> 5;   // 0..3 : i,f,g,o
    const int lane  = threadIdx.x & 31;
    const int half  = lane >> 4, ln = lane & 15;

    __shared__ float gl[4][16][16];

    const int hc   = slice * 16 + ln;           // hidden col (B lane col)
    const int gcol = gate * HIDDEN + hc;        // gate col 0..2047
    const int brow = mtile * 16 + ln;           // batch row (A lane row)

    const int t = tok[brow];
    const unsigned short* xrow    = emb + (long)t * HIDDEN;
    const unsigned short* hrow    = hprev + brow * HIDDEN;
    const unsigned short* wih_row = Wih + (long)gcol * HIDDEN;
    const unsigned short* whh_row = Whh + (long)gcol * HIDDEN;

    const float binit = bih[gcol] + bhh[gcol];
    v8f acc;
#pragma unroll
    for (int r = 0; r < 8; ++r) acc[r] = binit;

#pragma unroll 4
    for (int k0 = 0; k0 < HIDDEN; k0 += 32) {
        v16bf ax = load_a_bf(xrow, k0, half);
        v16bf bx = load_b_bf(wih_row, k0, half);
        acc = __builtin_amdgcn_wmma_f32_16x16x32_bf16(false, ax, false, bx,
                                                      (short)0, acc, false, false);
        v16bf ah = load_a_bf(hrow, k0, half);
        v16bf bh = load_b_bf(whh_row, k0, half);
        acc = __builtin_amdgcn_wmma_f32_16x16x32_bf16(false, ah, false, bh,
                                                      (short)0, acc, false, false);
    }
#pragma unroll
    for (int r = 0; r < 8; ++r) gl[gate][r + 8 * half][ln] = acc[r];
    __syncthreads();

    // pointwise LSTM cell: 256 elements, 128 threads
#pragma unroll
    for (int e = threadIdx.x; e < 256; e += 128) {
        const int m = e >> 4, n = e & 15;
        const float iv = gl[0][m][n], fv = gl[1][m][n];
        const float gv = gl[2][m][n], ov = gl[3][m][n];
        const long idx = (long)(mtile * 16 + m) * HIDDEN + slice * 16 + n;
        const float cold = cstate[idx];
        const float si = 1.f / (1.f + __expf(-iv));
        const float sf = 1.f / (1.f + __expf(-fv));
        const float so = 1.f / (1.f + __expf(-ov));
        const float cn = sf * cold + si * tanhf(gv);
        const float hn = so * tanhf(cn);
        cstate[idx] = cn;
        const unsigned short hb = f2bf(hn);
        hnext[idx] = hb;
        if (hdec) hdec[idx] = hb;
    }
}

// ---------------------------------------------------------------------------
// Batched decoder projection + fused per-chunk online-softmax partials.
// grid = (NCHUNK, MTILES), block = 256 (8 waves, 5 N-tiles each).
// ---------------------------------------------------------------------------
__global__ __launch_bounds__(256)
void logits_kernel(const unsigned short* __restrict__ Hd,  // 3136 x 512 bf16
                   const unsigned short* __restrict__ Wo,  // 32000 x 512 bf16
                   const float* __restrict__ bo,           // 32000
                   float* __restrict__ pmax,               // 3136 x NCHUNK
                   float* __restrict__ psum)               // 3136 x NCHUNK
{
    const int chunk = blockIdx.x;
    const int mtile = blockIdx.y;
    const int wv = threadIdx.x >> 5, lane = threadIdx.x & 31;
    const int half = lane >> 4, ln = lane & 15;

    __shared__ float smax[WVS][16], ssum[WVS][16];

    const unsigned short* arow = Hd + (long)(mtile * 16 + ln) * HIDDEN;

    const int tbase = chunk * CHUNK_T + wv * TPW;
    v8f acc[TPW];
    const unsigned short* wrow[TPW];
#pragma unroll
    for (int t = 0; t < TPW; ++t) {
        const int col = (tbase + t) * 16 + ln;
        const float b = bo[col];
#pragma unroll
        for (int r = 0; r < 8; ++r) acc[t][r] = b;
        wrow[t] = Wo + (long)col * HIDDEN;
    }
    for (int k0 = 0; k0 < HIDDEN; k0 += 32) {
        const v16bf a = load_a_bf(arow, k0, half);
#pragma unroll
        for (int t = 0; t < TPW; ++t) {
            const v16bf b = load_b_bf(wrow[t], k0, half);
            acc[t] = __builtin_amdgcn_wmma_f32_16x16x32_bf16(false, a, false, b,
                                                             (short)0, acc[t], false, false);
        }
    }
    // per-row (max, sum exp) within this wave's 80 columns; xor masks 1..8 stay
    // inside each 16-lane half, which holds one row's 16 columns.
#pragma unroll
    for (int r = 0; r < 8; ++r) {
        float mx = -3.0e38f;
#pragma unroll
        for (int t = 0; t < TPW; ++t) mx = fmaxf(mx, acc[t][r]);
#pragma unroll
        for (int s = 1; s < 16; s <<= 1) mx = fmaxf(mx, __shfl_xor(mx, s));
        float sm = 0.f;
#pragma unroll
        for (int t = 0; t < TPW; ++t) sm += __expf(acc[t][r] - mx);
#pragma unroll
        for (int s = 1; s < 16; s <<= 1) sm += __shfl_xor(sm, s);
        if (ln == r) { smax[wv][8 * half + r] = mx; ssum[wv][8 * half + r] = sm; }
    }
    __syncthreads();
    if (threadIdx.x < 16) {
        const int m = threadIdx.x;
        float gm = -3.0e38f;
#pragma unroll
        for (int w = 0; w < WVS; ++w) gm = fmaxf(gm, smax[w][m]);
        float gs = 0.f;
#pragma unroll
        for (int w = 0; w < WVS; ++w) gs += ssum[w][m] * __expf(smax[w][m] - gm);
        const int row = mtile * 16 + m;
        pmax[row * NCHUNK + chunk] = gm;
        psum[row * NCHUNK + chunk] = gs;
    }
}

// logit at target index per decoder row (tiny GEMV)
__global__ __launch_bounds__(256)
void ltgt_kernel(const unsigned short* __restrict__ Hd,
                 const unsigned short* __restrict__ Wo,
                 const float* __restrict__ bo,
                 const int* __restrict__ tgt,            // full 50x64
                 float* __restrict__ ltgt)
{
    const int row = blockIdx.x * blockDim.x + threadIdx.x;
    if (row >= DROWS) return;
    const int tstep = row / BATCH, b = row % BATCH;
    const int nxt = tgt[(tstep + 1) * BATCH + b];
    const unsigned short* h = Hd + (long)row * HIDDEN;
    const unsigned short* w = Wo + (long)nxt * HIDDEN;
    float s = bo[nxt];
    for (int k = 0; k < HIDDEN; ++k) s += bf2f(h[k]) * bf2f(w[k]);
    ltgt[row] = s;
}

// merge chunk partials -> per-row loss contribution
__global__ __launch_bounds__(256)
void rowloss_kernel(const float* __restrict__ pmax, const float* __restrict__ psum,
                    const float* __restrict__ ltgt, float* __restrict__ rowloss)
{
    const int row = blockIdx.x * blockDim.x + threadIdx.x;
    if (row >= DROWS) return;
    float M = -3.0e38f, S = 0.f;
    for (int c = 0; c < NCHUNK; ++c) {
        const float m = pmax[row * NCHUNK + c], s = psum[row * NCHUNK + c];
        const float nm = fmaxf(M, m);
        S = S * __expf(M - nm) + s * __expf(m - nm);
        M = nm;
    }
    const float lp = ltgt[row] - M - __logf(S);
    rowloss[row] = -lp * (1.f / BATCH);
}

// deterministic single-block final reduction
__global__ __launch_bounds__(256)
void final_sum_kernel(const float* __restrict__ rl, float* __restrict__ out)
{
    __shared__ float sm[256];
    float s = 0.f;
    for (int i = threadIdx.x; i < DROWS; i += 256) s += rl[i];
    sm[threadIdx.x] = s;
    __syncthreads();
    for (int w = 128; w > 0; w >>= 1) {
        if (threadIdx.x < w) sm[threadIdx.x] += sm[threadIdx.x + w];
        __syncthreads();
    }
    if (threadIdx.x == 0) out[0] = sm[0];
}

__global__ void convert_kernel(const float* __restrict__ src,
                               unsigned short* __restrict__ dst, int n)
{
    int i = blockIdx.x * blockDim.x + threadIdx.x;
    const int stride = gridDim.x * blockDim.x;
    for (; i < n; i += stride) dst[i] = f2bf(src[i]);
}

__global__ void init_kernel(unsigned short* h0, float* c)
{
    const int i = blockIdx.x * blockDim.x + threadIdx.x;
    if (i < BATCH * HIDDEN) { h0[i] = 0; c[i] = 0.f; }
}

extern "C" void kernel_launch(void* const* d_in, const int* in_sizes, int n_in,
                              void* d_out, int out_size, void* d_ws, size_t ws_size,
                              hipStream_t stream)
{
    (void)in_sizes; (void)n_in; (void)out_size; (void)ws_size;
    const int*   input_lines  = (const int*)  d_in[0];
    const int*   target_lines = (const int*)  d_in[1];
    const float* emb_in   = (const float*)d_in[2];
    const float* emb_tgt  = (const float*)d_in[3];
    const float* W_ih_enc = (const float*)d_in[4];
    const float* W_hh_enc = (const float*)d_in[5];
    const float* b_ih_enc = (const float*)d_in[6];
    const float* b_hh_enc = (const float*)d_in[7];
    const float* W_ih_dec = (const float*)d_in[8];
    const float* W_hh_dec = (const float*)d_in[9];
    const float* b_ih_dec = (const float*)d_in[10];
    const float* b_hh_dec = (const float*)d_in[11];
    const float* W_out    = (const float*)d_in[12];
    const float* b_out    = (const float*)d_in[13];
    float* out = (float*)d_out;

    // ---- workspace layout (256B aligned) ----
    char* ws = (char*)d_ws;
    size_t off = 0;
    auto alloc = [&](size_t bytes) -> char* {
        char* p = ws + off;
        off += (bytes + 255) & ~(size_t)255;
        return p;
    };
    const int GW = 4 * HIDDEN * HIDDEN;              // 2048*512
    const int EW = VOCAB * HIDDEN;                   // 32000*512
    unsigned short* wih_enc_bf = (unsigned short*)alloc((size_t)GW * 2);
    unsigned short* whh_enc_bf = (unsigned short*)alloc((size_t)GW * 2);
    unsigned short* wih_dec_bf = (unsigned short*)alloc((size_t)GW * 2);
    unsigned short* whh_dec_bf = (unsigned short*)alloc((size_t)GW * 2);
    unsigned short* wout_bf    = (unsigned short*)alloc((size_t)EW * 2);
    unsigned short* embin_bf   = (unsigned short*)alloc((size_t)EW * 2);
    unsigned short* embtg_bf   = (unsigned short*)alloc((size_t)EW * 2);
    unsigned short* hbf0       = (unsigned short*)alloc((size_t)BATCH * HIDDEN * 2);
    unsigned short* hbf1       = (unsigned short*)alloc((size_t)BATCH * HIDDEN * 2);
    float*          cbuf       = (float*)alloc((size_t)BATCH * HIDDEN * 4);
    unsigned short* hdec_bf    = (unsigned short*)alloc((size_t)DROWS * HIDDEN * 2);
    float*          pmax       = (float*)alloc((size_t)DROWS * NCHUNK * 4);
    float*          psum       = (float*)alloc((size_t)DROWS * NCHUNK * 4);
    float*          ltgt       = (float*)alloc((size_t)DROWS * 4);
    float*          rowloss    = (float*)alloc((size_t)DROWS * 4);

    // ---- one-time bf16 conversion of all matrix operands ----
    convert_kernel<<<512, 256, 0, stream>>>(W_ih_enc, wih_enc_bf, GW);
    convert_kernel<<<512, 256, 0, stream>>>(W_hh_enc, whh_enc_bf, GW);
    convert_kernel<<<512, 256, 0, stream>>>(W_ih_dec, wih_dec_bf, GW);
    convert_kernel<<<512, 256, 0, stream>>>(W_hh_dec, whh_dec_bf, GW);
    convert_kernel<<<4096, 256, 0, stream>>>(W_out,  wout_bf,  EW);
    convert_kernel<<<4096, 256, 0, stream>>>(emb_in,  embin_bf, EW);
    convert_kernel<<<4096, 256, 0, stream>>>(emb_tgt, embtg_bf, EW);
    init_kernel<<<(BATCH * HIDDEN + 255) / 256, 256, 0, stream>>>(hbf0, cbuf);

    unsigned short* hbf[2] = {hbf0, hbf1};
    int s = 0;
    // ---- encoder scan ----
    for (int t = 0; t < SRC_LEN; ++t, ++s) {
        lstm_step_kernel<<<dim3(32, 4), 128, 0, stream>>>(
            input_lines + t * BATCH, embin_bf, wih_enc_bf, whh_enc_bf,
            b_ih_enc, b_hh_enc, hbf[s & 1], hbf[(s + 1) & 1], cbuf,
            (unsigned short*)nullptr);
    }
    // ---- decoder scan (teacher forcing; stash h_t) ----
    for (int t = 0; t < DEC_STEPS; ++t, ++s) {
        lstm_step_kernel<<<dim3(32, 4), 128, 0, stream>>>(
            target_lines + t * BATCH, embtg_bf, wih_dec_bf, whh_dec_bf,
            b_ih_dec, b_hh_dec, hbf[s & 1], hbf[(s + 1) & 1], cbuf,
            hdec_bf + (size_t)t * BATCH * HIDDEN);
    }
    // ---- one batched projection (M=3136, N=32000, K=512) + fused softmax ----
    logits_kernel<<<dim3(NCHUNK, MTILES), 256, 0, stream>>>(hdec_bf, wout_bf, b_out,
                                                            pmax, psum);
    ltgt_kernel<<<(DROWS + 255) / 256, 256, 0, stream>>>(hdec_bf, wout_bf, b_out,
                                                         target_lines, ltgt);
    rowloss_kernel<<<(DROWS + 255) / 256, 256, 0, stream>>>(pmax, psum, ltgt, rowloss);
    final_sum_kernel<<<1, 256, 0, stream>>>(rowloss, out);
}
